// CCDPA_43473658970726
// MI455X (gfx1250) — matrix operand order repaired and checked by reading, not compiled, not run on hardware
//
#include <hip/hip_runtime.h>
#include <hip/hip_bf16.h>

typedef __attribute__((ext_vector_type(16))) __bf16 v16bf;
typedef __attribute__((ext_vector_type(8)))  float  v8f;

#define SPATIAL 32768   // D*W*H
#define KTOT    1024    // 4 modalities * 256 channels
#define BK      32
#define BN      64
#define KSTEPS  (KTOT / BK)           // 32
#define A_TILE_DW 4096                // 16 mtiles * 32 lanes * 8 dwords (16KB)
#define B_TILE_DW 1024                // 4 ntiles * 32 lanes * 8 dwords  (4KB)

__device__ __forceinline__ unsigned short f2bf(float f) {
  unsigned int u = __float_as_uint(f);
  unsigned int r = (u + 0x7FFFu + ((u >> 16) & 1u)) >> 16;
  return (unsigned short)r;
}

// pack bf16(hi)|bf16(lo) from two floats, truncation mode: one v_perm_b32
__device__ __forceinline__ unsigned int pack_bf16_pair(float lo, float hi) {
#if __has_builtin(__builtin_amdgcn_perm)
  return __builtin_amdgcn_perm(__float_as_uint(hi), __float_as_uint(lo),
                               0x07060302u);  // {hi[3:2], lo[3:2]}
#else
  return (__float_as_uint(hi) & 0xFFFF0000u) | (__float_as_uint(lo) >> 16);
#endif
}

// ---------------- Kernel 1: adaptive_avg_pool3d -> pooled[b][c][m][d] ----------------
__global__ void __launch_bounds__(256)
pool_kernel(const float* __restrict__ x0, const float* __restrict__ x1,
            const float* __restrict__ x2, const float* __restrict__ x3,
            float* __restrict__ pooled) {
  int wave = threadIdx.x >> 5;
  int lane = threadIdx.x & 31;
  int wg = blockIdx.x * 8 + wave;
  int d = wg & 31;
  int m = (wg >> 5) & 3;
  int c = (wg >> 7) & 255;
  int b = wg >> 15;
  const float* src = (m == 0) ? x0 : (m == 1) ? x1 : (m == 2) ? x2 : x3;
  const float4* p = (const float4*)(src + ((size_t)(b * 256 + c) * 32 + d) * 1024);
  float s = 0.f;
#pragma unroll
  for (int t = 0; t < 8; ++t) {
    float4 v = p[lane + 32 * t];
    s += v.x + v.y + v.z + v.w;
  }
#pragma unroll
  for (int off = 16; off >= 1; off >>= 1) s += __shfl_xor(s, off, 32);
  if (lane == 0) pooled[((b * 256 + c) * 4 + m) * 32 + d] = s * (1.0f / 1024.0f);
}

// ---------------- Kernel 2: attention -> W_eff pre-swizzled into WMMA-A fragment order
// W_eff global layout (dwords): [b][kstep(32)][mtile(16)][lane(32)][dw(8)]
__global__ void __launch_bounds__(256)
attn_kernel(const float* __restrict__ pooled,
            const float* __restrict__ Wq, const float* __restrict__ bq,
            const float* __restrict__ Wk, const float* __restrict__ bk,
            const float* __restrict__ Wc, const float* __restrict__ bc,
            float* __restrict__ bias_eff, unsigned int* __restrict__ weff_sw) {
  int b = blockIdx.x;
  int o = threadIdx.x;
  const float* pb = pooled + ((size_t)b * 256 + o) * 4 * 32;  // [m][d]
  float q[32];
#pragma unroll 4
  for (int e = 0; e < 32; ++e) {
    float acc = bq[e];
    for (int d = 0; d < 32; ++d) acc += pb[d] * Wq[e * 32 + d];
    q[e] = acc;
  }
  float logit[4];
  for (int m = 0; m < 4; ++m) {
    const float* pm = pb + m * 32;
    float dot = 0.f;
    for (int e = 0; e < 32; ++e) {
      float ke = bk[e];
      for (int d = 0; d < 32; ++d) ke += pm[d] * Wk[e * 32 + d];
      dot += q[e] * ke;
    }
    logit[m] = dot * 0.17677669529663687f;  // 1/sqrt(32)
  }
  float mx = fmaxf(fmaxf(logit[0], logit[1]), fmaxf(logit[2], logit[3]));
  float a[4], ssum = 0.f;
  for (int m = 0; m < 4; ++m) { a[m] = __expf(logit[m] - mx); ssum += a[m]; }
  float inv = 1.f / ssum;
  float be = 0.f;
  for (int m = 0; m < 4; ++m) { a[m] *= inv; be += a[m] * bc[m * 256 + o]; }
  bias_eff[b * 256 + o] = be;

  // emit W_eff, swizzled per ISA 16-bit A(16x32) layout:
  // lane g*16+mrow holds K = (j<8?0:16) + g*8 + (j&7) in dword j/2.
  // Weights use full RNE rounding (one-time cost, better accuracy).
  unsigned int* wsw = weff_sw + (size_t)b * (KSTEPS * A_TILE_DW);
  int mt = o >> 4, ml = o & 15;
  for (int m = 0; m < 4; ++m) {
    const float* wc = Wc + ((size_t)m * 256 + o) * 256;
    float am = a[m];
    for (int cp = 0; cp < 128; ++cp) {
      int K = m * 256 + cp * 2;     // even
      int kstep = K >> 5;
      int kl = K & 31;
      int g, dw;
      if (kl < 16) { g = kl >> 3;        dw = (kl & 7) >> 1; }
      else         { g = (kl - 16) >> 3; dw = 4 + ((kl & 7) >> 1); }
      unsigned int val = (unsigned int)f2bf(am * wc[cp * 2]) |
                         ((unsigned int)f2bf(am * wc[cp * 2 + 1]) << 16);
      wsw[((size_t)(kstep * 16 + mt) * 32 + g * 16 + ml) * 8 + dw] = val;
    }
  }
}

// ---------------- Kernel 3: fused conv-GEMM + bias via WMMA bf16 --------------------
// Per batch: Out[256 x 32768] = W_eff[256 x 1024] @ X[1024 x 32768] + bias.
// Block: 256 threads (8 wave32). Output tile 256 x 64.
// Software-pipelined: global loads of tile k+1 issue before the WMMAs of tile k.
struct StageRegs {
  uint4  a0, a1, a2, a3;   // A tile slice (already WMMA-swizzled)
  float4 f0, f1;           // B rows k, k+1 at 4 consecutive columns
};

__device__ __forceinline__ StageRegs stage_load(
    int kstep, const float* __restrict__ srcm, size_t brow, int s0,
    int tid, int kp, int n0, const unsigned int* __restrict__ Ab) {
  StageRegs st;
  const uint4* As = (const uint4*)(Ab + (size_t)kstep * A_TILE_DW);
  st.a0 = As[tid];
  st.a1 = As[tid + 256];
  st.a2 = As[tid + 512];
  st.a3 = As[tid + 768];
  // modality is uniform per k-step (m = kstep>>3); only c varies per thread
  int c = ((kstep & 7) << 5) + kp * 2;
  const float* p = srcm + brow + (size_t)c * SPATIAL + s0 + n0;
  st.f0 = *(const float4*)p;
  st.f1 = *(const float4*)(p + SPATIAL);
  return st;
}

__device__ __forceinline__ void stage_store(
    const StageRegs& st, int tid, int g, int dw, int n0,
    unsigned int* __restrict__ lA, unsigned int* __restrict__ lB) {
  uint4* lA4 = (uint4*)lA;
  lA4[tid]       = st.a0;
  lA4[tid + 256] = st.a1;
  lA4[tid + 512] = st.a2;
  lA4[tid + 768] = st.a3;
  float r0[4] = {st.f0.x, st.f0.y, st.f0.z, st.f0.w};
  float r1[4] = {st.f1.x, st.f1.y, st.f1.z, st.f1.w};
#pragma unroll
  for (int j = 0; j < 4; ++j) {
    int nl = n0 + j;
    lB[(((nl >> 4) * 32) + g * 16 + (nl & 15)) * 8 + dw] =
        pack_bf16_pair(r0[j], r1[j]);
  }
}

__global__ void __launch_bounds__(256)
fused_gemm_kernel(const float* __restrict__ x0, const float* __restrict__ x1,
                  const float* __restrict__ x2, const float* __restrict__ x3,
                  const unsigned int* __restrict__ weff_sw,
                  const float* __restrict__ bias_eff,
                  float* __restrict__ out) {
  __shared__ unsigned int ldsA[2][A_TILE_DW];  // 2 x 16KB
  __shared__ unsigned int ldsB[2][B_TILE_DW];  // 2 x 4KB

  int b  = blockIdx.y;
  int s0 = blockIdx.x * BN;
  int tid  = threadIdx.x;
  int lane = tid & 31;
  int wave = tid >> 5;

  int kp = tid >> 4;          // K-pair 0..15
  int n0 = (tid & 15) * 4;    // 4 consecutive columns
  int g  = (kp >> 2) & 1;
  int dw = ((kp >> 3) << 2) | (kp & 3);
  size_t brow = (size_t)b * 256 * SPATIAL;

  const unsigned int* Ab = weff_sw + (size_t)b * (KSTEPS * A_TILE_DW);

  v8f acc[2][4] = {};

  // scalar-uniform modality pointer per k-step
  auto pick = [&](int ks) -> const float* {
    int m = ks >> 3;
    return (m == 0) ? x0 : (m == 1) ? x1 : (m == 2) ? x2 : x3;
  };

  StageRegs st = stage_load(0, pick(0), brow, s0, tid, kp, n0, Ab);

  for (int step = 0; step < KSTEPS; ++step) {
    int cur = step & 1;
    stage_store(st, tid, g, dw, n0, ldsA[cur], ldsB[cur]);
    __syncthreads();
    if (step + 1 < KSTEPS)
      st = stage_load(step + 1, pick(step + 1), brow, s0, tid, kp, n0, Ab);

    v16bf bf[4];
#pragma unroll
    for (int nt = 0; nt < 4; ++nt)
      bf[nt] = *(const v16bf*)&ldsB[cur][(nt * 32 + lane) * 8];
#pragma unroll
    for (int mi = 0; mi < 2; ++mi) {
      v16bf af = *(const v16bf*)&ldsA[cur][((wave * 2 + mi) * 32 + lane) * 8];
#pragma unroll
      for (int nt = 0; nt < 4; ++nt)
        acc[mi][nt] = __builtin_amdgcn_wmma_f32_16x16x32_bf16(
            false, af, false, bf[nt], (short)0, acc[mi][nt], false, false);
    }
  }

  // ---- epilogue: add bias, store fp32 ----
  // C/D layout: lane<16 -> M=r, lane>=16 -> M=r+8; N = lane&15.
#pragma unroll
  for (int mi = 0; mi < 2; ++mi) {
    int mt = wave * 2 + mi;
    float bias[8];
#pragma unroll
    for (int r = 0; r < 8; ++r)
      bias[r] = bias_eff[b * 256 + mt * 16 + ((lane < 16) ? r : r + 8)];
#pragma unroll
    for (int nt = 0; nt < 4; ++nt) {
      int n = s0 + nt * 16 + (lane & 15);
#pragma unroll
      for (int r = 0; r < 8; ++r) {
        int o = mt * 16 + ((lane < 16) ? r : r + 8);
        out[((size_t)b * 256 + o) * SPATIAL + n] = acc[mi][nt][r] + bias[r];
      }
    }
  }
}

extern "C" void kernel_launch(void* const* d_in, const int* in_sizes, int n_in,
                              void* d_out, int out_size, void* d_ws, size_t ws_size,
                              hipStream_t stream) {
  const float* x0 = (const float*)d_in[0];
  const float* x1 = (const float*)d_in[1];
  const float* x2 = (const float*)d_in[2];
  const float* x3 = (const float*)d_in[3];
  const float* Wq = (const float*)d_in[4];
  const float* bq = (const float*)d_in[5];
  const float* Wk = (const float*)d_in[6];
  const float* bk = (const float*)d_in[7];
  const float* Wc = (const float*)d_in[8];
  const float* bc = (const float*)d_in[9];
  float* out = (float*)d_out;

  char* ws = (char*)d_ws;
  float*        pooled   = (float*)ws;                      // 512 KB
  float*        bias_eff = (float*)(ws + 524288);           // 4 KB
  unsigned int* weff_sw  = (unsigned int*)(ws + 528384);    // 2 MB (bf16 pairs, swizzled)

  // 1) pooling: 4*256*4*32 = 131072 waves -> 16384 blocks of 8 waves
  pool_kernel<<<16384, 256, 0, stream>>>(x0, x1, x2, x3, pooled);

  // 2) attention -> effective (pre-swizzled) bf16 weights
  attn_kernel<<<4, 256, 0, stream>>>(pooled, Wq, bq, Wk, bk, Wc, bc, bias_eff, weff_sw);

  // 3) fused GEMM: grid (32768/64 spatial tiles, 4 batches)
  dim3 grid(SPATIAL / BN, 4);
  fused_gemm_kernel<<<grid, 256, 0, stream>>>(x0, x1, x2, x3, weff_sw, bias_eff, out);
}
